// SpikeTransformerBlock_19894288515536
// MI455X (gfx1250) — matrix operand order repaired and checked by reading, not compiled, not run on hardware
//
#include <hip/hip_runtime.h>
#include <hip/hip_bf16.h>

typedef _Float16 h16;
typedef __attribute__((ext_vector_type(8)))  _Float16 v8h;
typedef __attribute__((ext_vector_type(16))) _Float16 v16h;
typedef __attribute__((ext_vector_type(8)))  float    v8f;

#define T_DIM 4
#define S_DIM 1024
#define D_DIM 768
#define FF_DIM 3072
#define M_DIM (T_DIM * S_DIM)   // 4096 rows for every GEMM
#define BN_EPS 1e-5f
#define SB_STRIDE 40            // halves per LDS column: 80B, 16B-aligned, bank-spread

// ---------------------------------------------------------------------------
// Weight convert + transpose: W[K,N] f32 -> WT[N,K] f16, so WMMA B-fragments
// are two contiguous 8-half k-runs per lane.
// ---------------------------------------------------------------------------
__global__ void conv_transpose_kernel(const float* __restrict__ W,
                                      h16* __restrict__ WT, int K, int N) {
  int idx = blockIdx.x * blockDim.x + threadIdx.x;
  if (idx >= K * N) return;
  int k = idx / N, n = idx % N;
  WT[(size_t)n * K + k] = (h16)W[idx];
}

// ---------------------------------------------------------------------------
// IF neuron on raw input x: v += x_t; spike = (v >= 1); hard reset.
// ---------------------------------------------------------------------------
__global__ void if_input_kernel(const float* __restrict__ x,
                                h16* __restrict__ sp) {
  int idx = blockIdx.x * blockDim.x + threadIdx.x;   // over S*D
  if (idx >= S_DIM * D_DIM) return;
  float v = 0.f;
#pragma unroll
  for (int t = 0; t < T_DIM; ++t) {
    v += x[(size_t)t * S_DIM * D_DIM + idx];
    float s = (v >= 1.f) ? 1.f : 0.f;
    sp[(size_t)t * S_DIM * D_DIM + idx] = (h16)s;
    v *= (1.f - s);
  }
}

// ---------------------------------------------------------------------------
// WMMA GEMM: C[M,N] = A[M,K] (f16 binary spikes) * BT[N,K]^T (f16 weights)
//            + optional bias[N] (f32).
// Block = 256 threads = 8 waves; block tile 128x64; wave tile 16x64.
// The shared 64x32 B tile is staged in LDS via a double-buffered
// global_load_async_to_lds_b128 pipeline (ASYNCcnt); each thread moves 16B
// per stage. Steady-state body is branch-free (last stage peeled). A
// sched_barrier(0) splits the 8 ds_load_b128s from the 4 WMMAs so the loads
// issue as one clause (single dscnt wait) and the WMMAs run back-to-back.
// K % 32 == 0, M % 128 == 0, N % 64 == 0 -> EXEC all ones at every WMMA.
// ---------------------------------------------------------------------------
__global__ __launch_bounds__(256)
void gemm_wmma_kernel(const h16* __restrict__ A, const h16* __restrict__ BT,
                      const float* __restrict__ bias, float* __restrict__ C,
                      int K, int N) {
  __shared__ h16 sB[2][64 * SB_STRIDE];

  const int tid  = threadIdx.x;
  const int lane = tid & 31;
  const int wave = tid >> 5;
  const int l15  = lane & 15;
  const int kb   = lane >> 4;           // which K-half this lane carries
  const int row0 = blockIdx.x * 128 + wave * 16;
  const int col0 = blockIdx.y * 64;

  // Per-thread async-copy job: 16B chunk cc of local column cn.
  const int cn = tid >> 2;              // 0..63
  const int cc = tid & 3;               // 0..3
  const h16* gsrc = BT + (size_t)(col0 + cn) * K + cc * 8;
  // Flat __shared__ pointer low 32 bits == LDS offset (ISA 10.2 aperture map).
  const unsigned ldst0 = (unsigned)(uintptr_t)&sB[0][cn * SB_STRIDE + cc * 8];
  const unsigned ldst1 = (unsigned)(uintptr_t)&sB[1][cn * SB_STRIDE + cc * 8];

  v8f acc[4] = {v8f{}, v8f{}, v8f{}, v8f{}};
  const h16* Arow = A + (size_t)(row0 + l15) * K;

  auto issue_copy = [&](int k0, unsigned ld) {
    const h16* g = gsrc + k0;
    asm volatile("global_load_async_to_lds_b128 %0, %1, off"
                 :: "v"(ld), "v"(g) : "memory");
  };

  // Load all four B fragments, fence the scheduler, then 4 WMMAs back-to-back.
  auto mma_stage = [&](const v8h& alo, const v8h& ahi, int buf) {
    v16h a;
#pragma unroll
    for (int i = 0; i < 8; ++i) { a[i] = alo[i]; a[8 + i] = ahi[i]; }

    v16h b[4];
#pragma unroll
    for (int nt = 0; nt < 4; ++nt) {
      const h16* bp = &sB[buf][(nt * 16 + l15) * SB_STRIDE + kb * 8];
      const v8h blo = *reinterpret_cast<const v8h*>(bp);
      const v8h bhi = *reinterpret_cast<const v8h*>(bp + 16);
#pragma unroll
      for (int i = 0; i < 8; ++i) { b[nt][i] = blo[i]; b[nt][8 + i] = bhi[i]; }
    }
#if __has_builtin(__builtin_amdgcn_sched_barrier)
    __builtin_amdgcn_sched_barrier(0);   // all ds loads before any WMMA
#endif
#pragma unroll
    for (int nt = 0; nt < 4; ++nt) {
      acc[nt] = __builtin_amdgcn_wmma_f32_16x16x32_f16(
          false, a, false, b[nt], (short)0, acc[nt], false, false);
    }
  };

  const int nstages = K / 32;

  // Prologue: stage 0 -> buffer 0.
  issue_copy(0, ldst0);

  // Steady state: branch-free; next copy in flight while computing current.
  for (int s = 0; s < nstages - 1; ++s) {
    const int k0 = s * 32;
    const int buf = s & 1;

    // A fragment (16x32 f16): lane l15 = row M; kb selects K sub-blocks
    // {kb*8..+7} and {16+kb*8..+7} -> two 16B global loads (overlap copies).
    const v8h alo = *reinterpret_cast<const v8h*>(Arow + k0 + kb * 8);
    const v8h ahi = *reinterpret_cast<const v8h*>(Arow + k0 + 16 + kb * 8);
    __builtin_prefetch(Arow + k0 + 32, 0, 0);

    issue_copy(k0 + 32, buf ? ldst0 : ldst1);
    asm volatile("s_wait_asynccnt 0x1" ::: "memory");  // current tile landed
    __syncthreads();                                    // visible to all waves

    mma_stage(alo, ahi, buf);
    __syncthreads();   // everyone done reading buf before it is rewritten
  }

  // Epilogue stage (peeled): drain the pipeline completely.
  {
    const int k0 = (nstages - 1) * 32;
    const int buf = (nstages - 1) & 1;
    const v8h alo = *reinterpret_cast<const v8h*>(Arow + k0 + kb * 8);
    const v8h ahi = *reinterpret_cast<const v8h*>(Arow + k0 + 16 + kb * 8);
    asm volatile("s_wait_asynccnt 0x0" ::: "memory");
    __syncthreads();
    mma_stage(alo, ahi, buf);
  }

  // Epilogue: C/D layout -> VGPR r, lane: M = r + 8*kb, N = l15.
#pragma unroll
  for (int nt = 0; nt < 4; ++nt) {
    const int col = col0 + nt * 16 + l15;
    const float bv = bias ? bias[col] : 0.f;
#pragma unroll
    for (int r = 0; r < 8; ++r) {
      const int row = row0 + r + 8 * kb;
      C[(size_t)row * N + col] = acc[nt][r] + bv;
    }
  }
}

// ---------------------------------------------------------------------------
// BatchNorm1d(S) over (T, C) per channel s, then IF neuron over T.
// ---------------------------------------------------------------------------
__global__ __launch_bounds__(256)
void bn_if_kernel(const float* __restrict__ in, h16* __restrict__ sp, int C) {
  const int s = blockIdx.x;
  const int tid = threadIdx.x;
  __shared__ float r1[256], r2[256];
  const int TC = T_DIM * C;
  float s1 = 0.f, s2 = 0.f;
  for (int i = tid; i < TC; i += 256) {
    int t = i / C, c = i % C;
    float v = in[(size_t)t * S_DIM * C + (size_t)s * C + c];
    s1 += v; s2 += v * v;
  }
  r1[tid] = s1; r2[tid] = s2;
  __syncthreads();
  for (int off = 128; off > 0; off >>= 1) {
    if (tid < off) { r1[tid] += r1[tid + off]; r2[tid] += r2[tid + off]; }
    __syncthreads();
  }
  const float mean = r1[0] / (float)TC;
  const float var  = r2[0] / (float)TC - mean * mean;
  const float rs   = rsqrtf(var + BN_EPS);

  for (int c = tid; c < C; c += 256) {
    float v = 0.f;
#pragma unroll
    for (int t = 0; t < T_DIM; ++t) {
      const size_t o = (size_t)t * S_DIM * C + (size_t)s * C + c;
      v += (in[o] - mean) * rs;
      float spk = (v >= 1.f) ? 1.f : 0.f;
      sp[o] = (h16)spk;
      v *= (1.f - spk);
    }
  }
}

// ---------------------------------------------------------------------------
// SDSA core: QK[t,d] = sum_s Q[t,s,d]*K[t,s,d], then IF over t (talking heads).
// ---------------------------------------------------------------------------
__global__ __launch_bounds__(256)
void qk_if_kernel(const h16* __restrict__ Qs, const h16* __restrict__ Ks,
                  h16* __restrict__ qksp) {
  const int tid = threadIdx.x;
  const int dL = tid & 15;
  const int sL = tid >> 4;
  const int d = blockIdx.x * 16 + dL;
  __shared__ float red[256];
  float v = 0.f;                       // IF state (valid on sL==0 threads)
  for (int t = 0; t < T_DIM; ++t) {
    float partial = 0.f;
    for (int s = sL; s < S_DIM; s += 16) {
      const size_t o = (size_t)t * S_DIM * D_DIM + (size_t)s * D_DIM + d;
      partial += (float)Qs[o] * (float)Ks[o];
    }
    red[sL * 16 + dL] = partial;
    __syncthreads();
    if (sL == 0) {
      float tot = 0.f;
#pragma unroll
      for (int j = 0; j < 16; ++j) tot += red[j * 16 + dL];
      v += tot;
      float spk = (v >= 1.f) ? 1.f : 0.f;
      qksp[t * D_DIM + d] = (h16)spk;
      v *= (1.f - spk);
    }
    __syncthreads();
  }
}

// QKV[t,s,d] = V[t,s,d] * qkspike[t,d]   (binary x binary -> binary f16)
__global__ void qkv_mul_kernel(const h16* __restrict__ Vs,
                               const h16* __restrict__ qksp,
                               h16* __restrict__ qkv) {
  int idx = blockIdx.x * blockDim.x + threadIdx.x;
  if (idx >= T_DIM * S_DIM * D_DIM) return;
  int t = idx / (S_DIM * D_DIM);
  int d = idx % D_DIM;
  qkv[idx] = (h16)((float)Vs[idx] * (float)qksp[t * D_DIM + d]);
}

// ---------------------------------------------------------------------------
// attn = BN(QKV@wo); h = x + attn; fspike = IF(h).
// ---------------------------------------------------------------------------
__global__ __launch_bounds__(256)
void bn_attn_kernel(const float* __restrict__ olin, const float* __restrict__ x,
                    float* __restrict__ h, h16* __restrict__ fsp) {
  const int s = blockIdx.x;
  const int tid = threadIdx.x;
  __shared__ float r1[256], r2[256];
  const int TC = T_DIM * D_DIM;
  float s1 = 0.f, s2 = 0.f;
  for (int i = tid; i < TC; i += 256) {
    int t = i / D_DIM, c = i % D_DIM;
    float v = olin[(size_t)t * S_DIM * D_DIM + (size_t)s * D_DIM + c];
    s1 += v; s2 += v * v;
  }
  r1[tid] = s1; r2[tid] = s2;
  __syncthreads();
  for (int off = 128; off > 0; off >>= 1) {
    if (tid < off) { r1[tid] += r1[tid + off]; r2[tid] += r2[tid + off]; }
    __syncthreads();
  }
  const float mean = r1[0] / (float)TC;
  const float var  = r2[0] / (float)TC - mean * mean;
  const float rs   = rsqrtf(var + BN_EPS);

  for (int c = tid; c < D_DIM; c += 256) {
    float v = 0.f;
#pragma unroll
    for (int t = 0; t < T_DIM; ++t) {
      const size_t o = (size_t)t * S_DIM * D_DIM + (size_t)s * D_DIM + c;
      const float hv = x[o] + (olin[o] - mean) * rs;
      h[o] = hv;
      v += hv;
      float spk = (v >= 1.f) ? 1.f : 0.f;
      fsp[o] = (h16)spk;
      v *= (1.f - spk);
    }
  }
}

// out = h + BN(f2lin).
__global__ __launch_bounds__(256)
void bn_out_kernel(const float* __restrict__ f2lin, const float* __restrict__ h,
                   float* __restrict__ out) {
  const int s = blockIdx.x;
  const int tid = threadIdx.x;
  __shared__ float r1[256], r2[256];
  const int TC = T_DIM * D_DIM;
  float s1 = 0.f, s2 = 0.f;
  for (int i = tid; i < TC; i += 256) {
    int t = i / D_DIM, c = i % D_DIM;
    float v = f2lin[(size_t)t * S_DIM * D_DIM + (size_t)s * D_DIM + c];
    s1 += v; s2 += v * v;
  }
  r1[tid] = s1; r2[tid] = s2;
  __syncthreads();
  for (int off = 128; off > 0; off >>= 1) {
    if (tid < off) { r1[tid] += r1[tid + off]; r2[tid] += r2[tid + off]; }
    __syncthreads();
  }
  const float mean = r1[0] / (float)TC;
  const float var  = r2[0] / (float)TC - mean * mean;
  const float rs   = rsqrtf(var + BN_EPS);

  for (int i = tid; i < TC; i += 256) {
    int t = i / D_DIM, c = i % D_DIM;
    const size_t o = (size_t)t * S_DIM * D_DIM + (size_t)s * D_DIM + c;
    out[o] = h[o] + (f2lin[o] - mean) * rs;
  }
}

// ---------------------------------------------------------------------------
extern "C" void kernel_launch(void* const* d_in, const int* in_sizes, int n_in,
                              void* d_out, int out_size, void* d_ws, size_t ws_size,
                              hipStream_t stream) {
  const float* x  = (const float*)d_in[0];
  const float* wq = (const float*)d_in[1];
  const float* wk = (const float*)d_in[2];
  const float* wv = (const float*)d_in[3];
  const float* wo = (const float*)d_in[4];
  const float* w1 = (const float*)d_in[5];
  const float* b1 = (const float*)d_in[6];
  const float* w2 = (const float*)d_in[7];
  const float* b2 = (const float*)d_in[8];
  float* out = (float*)d_out;

  // Static scratch carve-out (256B aligned).
  char* ws = (char*)d_ws;
  size_t off = 0;
  auto take = [&](size_t bytes) -> char* {
    char* p = ws + off;
    off += (bytes + 255) & ~(size_t)255;
    return p;
  };
  h16* wqT  = (h16*)take((size_t)D_DIM * D_DIM * sizeof(h16));
  h16* wkT  = (h16*)take((size_t)D_DIM * D_DIM * sizeof(h16));
  h16* wvT  = (h16*)take((size_t)D_DIM * D_DIM * sizeof(h16));
  h16* woT  = (h16*)take((size_t)D_DIM * D_DIM * sizeof(h16));
  h16* w1T  = (h16*)take((size_t)D_DIM * FF_DIM * sizeof(h16));   // [FF, D]
  h16* w2T  = (h16*)take((size_t)FF_DIM * D_DIM * sizeof(h16));   // [D, FF]
  h16* tmp  = (h16*)take((size_t)M_DIM * D_DIM * sizeof(h16));    // input spikes
  float* lin = (float*)take((size_t)M_DIM * FF_DIM * sizeof(float)); // shared GEMM out
  h16* Qs   = (h16*)take((size_t)M_DIM * D_DIM * sizeof(h16));
  h16* Ks   = (h16*)take((size_t)M_DIM * D_DIM * sizeof(h16));
  h16* Vs   = (h16*)take((size_t)M_DIM * D_DIM * sizeof(h16));
  h16* qksp = (h16*)take((size_t)T_DIM * D_DIM * sizeof(h16));
  h16* qkvm = (h16*)take((size_t)M_DIM * D_DIM * sizeof(h16));
  float* h  = (float*)take((size_t)M_DIM * D_DIM * sizeof(float));
  h16* fsp  = (h16*)take((size_t)M_DIM * D_DIM * sizeof(h16));
  h16* f2sp = (h16*)take((size_t)M_DIM * FF_DIM * sizeof(h16));

  const int ETB = 256;
  const int nDD = D_DIM * D_DIM, nDF = D_DIM * FF_DIM;

  // 1) Weight convert + transpose (f32 -> f16^T).
  conv_transpose_kernel<<<(nDD + ETB - 1) / ETB, ETB, 0, stream>>>(wq, wqT, D_DIM, D_DIM);
  conv_transpose_kernel<<<(nDD + ETB - 1) / ETB, ETB, 0, stream>>>(wk, wkT, D_DIM, D_DIM);
  conv_transpose_kernel<<<(nDD + ETB - 1) / ETB, ETB, 0, stream>>>(wv, wvT, D_DIM, D_DIM);
  conv_transpose_kernel<<<(nDD + ETB - 1) / ETB, ETB, 0, stream>>>(wo, woT, D_DIM, D_DIM);
  conv_transpose_kernel<<<(nDF + ETB - 1) / ETB, ETB, 0, stream>>>(w1, w1T, D_DIM, FF_DIM);
  conv_transpose_kernel<<<(nDF + ETB - 1) / ETB, ETB, 0, stream>>>(w2, w2T, FF_DIM, D_DIM);

  // 2) tmp = IF(x) as binary f16.
  if_input_kernel<<<(S_DIM * D_DIM + ETB - 1) / ETB, ETB, 0, stream>>>(x, tmp);

  dim3 gD(M_DIM / 128, D_DIM / 64);   // GEMMs with N = 768
  dim3 gF(M_DIM / 128, FF_DIM / 64);  // GEMM  with N = 3072

  // 3) Q/K/V: GEMM -> BN + IF -> binary spikes.
  gemm_wmma_kernel<<<gD, ETB, 0, stream>>>(tmp, wqT, nullptr, lin, D_DIM, D_DIM);
  bn_if_kernel<<<S_DIM, ETB, 0, stream>>>(lin, Qs, D_DIM);
  gemm_wmma_kernel<<<gD, ETB, 0, stream>>>(tmp, wkT, nullptr, lin, D_DIM, D_DIM);
  bn_if_kernel<<<S_DIM, ETB, 0, stream>>>(lin, Ks, D_DIM);
  gemm_wmma_kernel<<<gD, ETB, 0, stream>>>(tmp, wvT, nullptr, lin, D_DIM, D_DIM);
  bn_if_kernel<<<S_DIM, ETB, 0, stream>>>(lin, Vs, D_DIM);

  // 4) SDSA: QK reduce over S + talking-heads IF; QKV = V * spike.
  qk_if_kernel<<<D_DIM / 16, ETB, 0, stream>>>(Qs, Ks, qksp);
  qkv_mul_kernel<<<(M_DIM * D_DIM + ETB - 1) / ETB, ETB, 0, stream>>>(Vs, qksp, qkvm);

  // 5) attn projection; h = x + BN(.); fspike = IF(h).
  gemm_wmma_kernel<<<gD, ETB, 0, stream>>>(qkvm, woT, nullptr, lin, D_DIM, D_DIM);
  bn_attn_kernel<<<S_DIM, ETB, 0, stream>>>(lin, x, h, fsp);

  // 6) FFN: (f@w1+b1) -> BN+IF -> (f@w2+b2) -> BN -> +h.
  gemm_wmma_kernel<<<gF, ETB, 0, stream>>>(fsp, w1T, b1, lin, D_DIM, FF_DIM);
  bn_if_kernel<<<S_DIM, ETB, 0, stream>>>(lin, f2sp, FF_DIM);
  gemm_wmma_kernel<<<gD, ETB, 0, stream>>>(f2sp, w2T, b2, lin, FF_DIM, D_DIM);
  bn_out_kernel<<<S_DIM, ETB, 0, stream>>>(lin, h, out);

  (void)in_sizes; (void)n_in; (void)out_size; (void)ws_size;
}
